// StageNetLayer_30923764531270
// MI455X (gfx1250) — matrix operand order repaired and checked by reading, not compiled, not run on hardware
//
#include <hip/hip_runtime.h>
#include <hip/hip_bf16.h>

typedef _Float16 h16;
typedef __attribute__((ext_vector_type(16))) _Float16 v16h;
typedef __attribute__((ext_vector_type(8)))  float    v8f;
typedef int v4i __attribute__((vector_size(16)));   // matches builtin's V4i param

#define AS1 __attribute__((address_space(1)))
#define AS3 __attribute__((address_space(3)))

#if defined(__gfx1250__) && __has_builtin(__builtin_amdgcn_global_load_async_to_lds_b128) && __has_builtin(__builtin_amdgcn_s_wait_asynccnt)
#define HAS_ASYNC 1
#else
#define HAS_ASYNC 0
#endif

#define B_   256
#define T_   256
#define D_   256
#define L_   3
#define CH_  128
#define H_   384
#define G_   1542
#define K_   10
#define HS_  64
#define KC_  672      // combined K: 257 (x,1) + 385 (h,1) = 642, padded to 672
#define KC2_ 3840     // conv K = H_*K_
#define XO_STRIDE 1552
#define MTILE 16
#define NT1  97       // ceil(1542/16)
#define NT2  24       // 384/16
#define THREADS 512
#define NWAVE 16

// ---- shared memory layout (bytes) ----
#define SM_X       0
#define SM_X_SZ    (MTILE*XO_STRIDE*4)            // 99328: xo OR {conv,mh,theme,t1}
#define SM_ACT     (SM_X + SM_X_SZ)
#define SM_ACT_SZ  (MTILE*KC_*2)                  // 21504
#define SM_TMPH    (SM_ACT + SM_ACT_SZ)
#define SM_TMPH_SZ (K_*MTILE*H_*2)                // 122880
#define SM_C       (SM_TMPH + SM_TMPH_SZ)
#define SM_C_SZ    (MTILE*H_*4)
#define SM_H       (SM_C + SM_C_SZ)
#define SM_H_SZ    (MTILE*H_*4)
#define SM_DIS     (SM_H + SM_H_SZ)
#define SM_DIS_SZ  (K_*MTILE*4)
#define SM_FM      (SM_DIS + SM_DIS_SZ)
#define SM_FM_SZ   (MTILE*L_*4)
#define SM_IM      (SM_FM + SM_FM_SZ)
#define SM_IM_SZ   (MTILE*L_*4)
#define SM_LD      (SM_IM + SM_IM_SZ)
#define SM_LD_SZ   (MTILE*K_*4)
#define SM_IDX     (SM_LD + SM_LD_SZ)
#define SM_IDX_SZ  (MTILE*4)
#define SM_XSTG    (SM_IDX + SM_IDX_SZ)           // x_t staging (async DMA target)
#define SM_XSTG_SZ (MTILE*D_*4)                   // 16384
#define SM_TOTAL   (SM_XSTG + SM_XSTG_SZ)
// aliases inside SM_X used after GEMM1 consumers are done:
#define SMX_CONV   0
#define SMX_MH     (MTILE*H_*4)
#define SMX_THEME  (2*MTILE*H_*4)
#define SMX_T1     (3*MTILE*H_*4)

static_assert(SM_TOTAL <= 320*1024, "LDS budget (320KB/WGP on CDNA5)");
static_assert((SM_XSTG & 15) == 0, "staging must be 16B aligned for b128 DMA");

// ---- WMMA fragment loaders (contiguous k = 16h + s order, shared by A and B) ----
// A 16x32: lane<16 holds K k0..k0+15; lane>=16 holds K k0+16..k0+31.
__device__ __forceinline__ v16h load_frag_a_lds(const h16* row, int k0, int lane) {
    int hb = (lane >> 4) << 4;
    union { uint4 u[2]; v16h v; } r;
    r.u[0] = *(const uint4*)(row + k0 + hb);
    r.u[1] = *(const uint4*)(row + k0 + hb + 8);
    return r.v;
}
// B 32x16 from K-major weights WT[n][k]: lanes 0-15 hold K k0..k0+15 of col n,
// lanes 16-31 hold K k0+16..k0+31 (contiguous 32B per lane).
__device__ __forceinline__ v16h load_frag_b_glob(const h16* __restrict__ W, int rowstride,
                                                 int n, int k0, int lane) {
    const h16* p = W + (size_t)n * rowstride + k0 + ((lane >> 4) << 4);
    union { uint4 u[2]; v16h v; } r;
    r.u[0] = ((const uint4*)p)[0];
    r.u[1] = ((const uint4*)p)[1];
    return r.v;
}

// ---- async DMA of x[.,t,:] tile into LDS staging (overlaps the WMMA GEMMs) ----
__device__ __forceinline__ void issue_x_async(const float* __restrict__ x, float* stgS,
                                              int bbase, int t, int tid) {
#if HAS_ASYNC
    #pragma unroll
    for (int c = 0; c < 2; ++c) {
        int chunk = tid + c * THREADS;                 // 1024 x 16B chunks
        int m = chunk >> 6, col = (chunk & 63) << 2;   // 64 chunks per row
        const float* g = x + ((size_t)(bbase + m) * T_ + t) * D_ + col;
        float*       l = stgS + m * D_ + col;
        __builtin_amdgcn_global_load_async_to_lds_b128(
            (AS1 v4i*)(uintptr_t)g, (AS3 v4i*)l, 0, 0);
    }
#else
    (void)x; (void)stgS; (void)bbase; (void)t; (void)tid;
#endif
}

// ---- weight conversion: WT1[g][k] f16, K-major, biases folded into ones-rows ----
__global__ void k_build_wt1(const float* __restrict__ Wk, const float* __restrict__ bk,
                            const float* __restrict__ Wr, const float* __restrict__ br,
                            h16* __restrict__ WT1) {
    int i = blockIdx.x * blockDim.x + threadIdx.x;
    if (i >= G_ * KC_) return;
    int g = i / KC_, k = i - g * KC_;
    float v;
    if      (k < D_)   v = Wk[(size_t)k * G_ + g];
    else if (k == D_)  v = Wk[(size_t)D_ * G_ + g] + bk[g];
    else if (k < 641)  v = Wr[(size_t)(k - 257) * G_ + g];
    else if (k == 641) v = Wr[(size_t)384 * G_ + g] + br[g];
    else               v = 0.f;
    WT1[i] = (h16)v;
}

// WT2[o][k*H+h] = Wc[o][h][k]  (conv-as-GEMM weight, K-major f16)
__global__ void k_build_wt2(const float* __restrict__ Wc, h16* __restrict__ WT2) {
    int i = blockIdx.x * blockDim.x + threadIdx.x;
    if (i >= H_ * KC2_) return;
    int o = i / KC2_, j = i - o * KC2_;
    int k = j / H_, hh = j - k * H_;
    WT2[i] = (h16)Wc[((size_t)o * H_ + hh) * K_ + k];
}

// ---- fused persistent recurrence: 16 WGs x 16 batch rows, all T in-kernel ----
__global__ __launch_bounds__(THREADS)
void k_stagenet(const float* __restrict__ x, const int* __restrict__ lengths,
                const h16* __restrict__ WT1, const h16* __restrict__ WT2,
                const float* __restrict__ Ws, const float* __restrict__ bs,
                const float* __restrict__ Wrs, const float* __restrict__ brs,
                const float* __restrict__ bc, float* __restrict__ out) {
    extern __shared__ char smem[];
    h16*   act    = (h16*)  (smem + SM_ACT);
    h16*   tmph   = (h16*)  (smem + SM_TMPH);
    float* xo     = (float*)(smem + SM_X);
    float* convS  = (float*)(smem + SM_X + SMX_CONV);
    float* mhS    = (float*)(smem + SM_X + SMX_MH);
    float* themeS = (float*)(smem + SM_X + SMX_THEME);
    float* t1S    = (float*)(smem + SM_X + SMX_T1);
    float* cS     = (float*)(smem + SM_C);
    float* hS     = (float*)(smem + SM_H);
    float* disS   = (float*)(smem + SM_DIS);
    float* fmS    = (float*)(smem + SM_FM);
    float* imS    = (float*)(smem + SM_IM);
    float* ldS    = (float*)(smem + SM_LD);
    int*   idxS   = (int*)  (smem + SM_IDX);
    float* stgS   = (float*)(smem + SM_XSTG);

    const int tid = threadIdx.x, lane = tid & 31, wid = tid >> 5;
    const int bbase = blockIdx.x * MTILE;
    float* outLast = out;
    float* outSeq  = out + (size_t)B_ * H_;

    for (int e = tid; e < MTILE * H_; e += THREADS) { cS[e] = 0.f; hS[e] = 0.f; }
    for (int e = tid; e < K_ * MTILE * H_; e += THREADS) tmph[e] = (h16)0.f;
    for (int e = tid; e < K_ * MTILE; e += THREADS) disS[e] = 0.f;
    if (tid < MTILE) {
        int len = lengths[bbase + tid];
        len = len < 1 ? 1 : (len > T_ ? T_ : len);
        idxS[tid] = len - 1;
    }
    __syncthreads();
#if HAS_ASYNC
    issue_x_async(x, stgS, bbase, 0, tid);   // prime the pipeline with x_0
#endif

    for (int t = 0; t < T_; ++t) {
        // ---- retire async x_t DMA, then build activation [x_t,1,h,1,pad] as f16 ----
#if HAS_ASYNC
        __builtin_amdgcn_s_wait_asynccnt(0);
        __syncthreads();
#endif
        for (int e = tid; e < MTILE * KC_; e += THREADS) {
            int m = e / KC_, k = e - m * KC_;
            float v;
#if HAS_ASYNC
            if      (k < D_)   v = stgS[m * D_ + k];
#else
            if      (k < D_)   v = x[((size_t)(bbase + m) * T_ + t) * D_ + k];
#endif
            else if (k == D_)  v = 1.f;
            else if (k < 641)  v = hS[m * H_ + (k - 257)];
            else if (k == 641) v = 1.f;
            else               v = 0.f;
            act[m * KC_ + k] = (h16)v;
        }
        __syncthreads();
#if HAS_ASYNC
        if (t + 1 < T_) issue_x_async(x, stgS, bbase, t + 1, tid);  // overlap with GEMMs
#endif

        // ---- GEMM1: xo(16x1542) = act(16x672) @ W(672x1542), f16 WMMA ----
        {
            const h16* arow = act + (lane & 15) * KC_;
            for (int tile = wid; tile < NT1; tile += NWAVE) {
                int n0 = tile * 16;
                int n  = n0 + (lane & 15); if (n > G_ - 1) n = G_ - 1;
                __builtin_prefetch(WT1 + (size_t)n * KC_, 0, 1);
                v8f c = {};
                #pragma unroll
                for (int k0 = 0; k0 < KC_; k0 += 32) {
                    v16h a = load_frag_a_lds(arow, k0, lane);
                    v16h b = load_frag_b_glob(WT1, KC_, n, k0, lane);
                    c = __builtin_amdgcn_wmma_f32_16x16x32_f16(false, a, false, b,
                                                               (short)0, c, false, false);
                }
                int mrow = (lane >> 4) << 3, ncol = lane & 15;
                #pragma unroll
                for (int r = 0; r < 8; ++r)
                    xo[(mrow + r) * XO_STRIDE + n0 + ncol] = c[r];
            }
        }
        __syncthreads();

        // ---- master gates (cumax L=3), dis ring, ld softmax ----
        if (tid < MTILE) {
            int m = tid;
            const float* xr = xo + m * XO_STRIDE;
            float z0 = xr[0], z1 = xr[1], z2 = xr[2];
            float mx = fmaxf(z0, fmaxf(z1, z2));
            float e0 = __expf(z0 - mx), e1 = __expf(z1 - mx), e2 = __expf(z2 - mx);
            float inv = 1.f / (e0 + e1 + e2);
            float p0 = e0 * inv, p1 = e1 * inv, p2 = e2 * inv;
            fmS[m * 3 + 0] = p0; fmS[m * 3 + 1] = p0 + p1; fmS[m * 3 + 2] = p0 + p1 + p2;
            z0 = xr[3]; z1 = xr[4]; z2 = xr[5];
            mx = fmaxf(z0, fmaxf(z1, z2));
            e0 = __expf(z0 - mx); e1 = __expf(z1 - mx); e2 = __expf(z2 - mx);
            inv = 1.f / (e0 + e1 + e2);
            p0 = e0 * inv; p1 = e1 * inv; p2 = e2 * inv;
            imS[m * 3 + 0] = p0 + p1 + p2; imS[m * 3 + 1] = p1 + p2; imS[m * 3 + 2] = p2;
            float cur = 1.f - (fmS[m * 3 + 0] + fmS[m * 3 + 1] + fmS[m * 3 + 2]) * (1.f / 3.f);
            disS[(t % K_) * MTILE + m] = cur;
            // ld = softmax(cumsum(dis over chronological window))
            float cum[K_]; float cacc = 0.f, cmx = -1e30f;
            for (int k = 0; k < K_; ++k) {
                cacc += disS[((t + 1 + k) % K_) * MTILE + m];
                cum[k] = cacc; cmx = fmaxf(cmx, cacc);
            }
            float ssum = 0.f;
            for (int k = 0; k < K_; ++k) { cum[k] = __expf(cum[k] - cmx); ssum += cum[k]; }
            float sinv = 1.f / ssum;
            for (int k = 0; k < K_; ++k) ldS[m * K_ + k] = cum[k] * sinv;
        }
        __syncthreads();

        // ---- ON-LSTM cell update (fp32), push h into f16 ring ----
        for (int e = tid; e < MTILE * H_; e += THREADS) {
            int m = e / H_, hh = e - m * H_;
            int l = hh / CH_;
            const float* xr = xo + m * XO_STRIDE;
            float f   = 1.f / (1.f + __expf(-xr[6 + hh]));
            float ii  = 1.f / (1.f + __expf(-xr[6 + H_ + hh]));
            float oo  = 1.f / (1.f + __expf(-xr[6 + 2 * H_ + hh]));
            float ci  = tanhf(xr[6 + 3 * H_ + hh]);
            float fmv = fmS[m * 3 + l], imv = imS[m * 3 + l], ovv = fmv * imv;
            float cl  = cS[e];
            float cn  = ovv * (f * cl + ii * ci) + (fmv - ovv) * cl + (imv - ovv) * ci;
            float hn  = oo * tanhf(cn);
            cS[e] = cn; hS[e] = hn;
            tmph[((t % K_) * MTILE + m) * H_ + hh] = (h16)hn;
        }
        __syncthreads();

        // ---- mh = mean_k(local_h) (fp32)  +  conv GEMM (16x3840 @ 3840x384, WMMA) ----
        for (int e = tid; e < MTILE * H_; e += THREADS) {
            int m = e / H_, hh = e - m * H_;
            float acc = 0.f;
            #pragma unroll
            for (int k = 0; k < K_; ++k)
                acc += (float)tmph[(((t + 1 + k) % K_) * MTILE + m) * H_ + hh] * ldS[m * K_ + k];
            mhS[e] = acc * (1.f / K_);
        }
        {
            int mA = lane & 15;
            for (int tile = wid; tile < NT2; tile += NWAVE) {
                int n0 = tile * 16;
                int n  = n0 + (lane & 15);
                __builtin_prefetch(WT2 + (size_t)n * KC2_, 0, 1);
                v8f c = {};
                for (int kk = 0; kk < K_; ++kk) {
                    int ridx = (t + 1 + kk) % K_;
                    const h16* trow = tmph + (ridx * MTILE + mA) * H_;
                    h16 lscale = (h16)ldS[mA * K_ + kk];
                    #pragma unroll
                    for (int hb0 = 0; hb0 < H_; hb0 += 32) {
                        v16h a = load_frag_a_lds(trow, hb0, lane);
                        a = a * lscale;  // fold ld[b,k] into A fragment
                        v16h b = load_frag_b_glob(WT2, KC2_, n, kk * H_ + hb0, lane);
                        c = __builtin_amdgcn_wmma_f32_16x16x32_f16(false, a, false, b,
                                                                   (short)0, c, false, false);
                    }
                }
                int mrow = (lane >> 4) << 3, ncol = lane & 15;
                float bcv = bc[n0 + ncol];
                #pragma unroll
                for (int r = 0; r < 8; ++r)
                    convS[(mrow + r) * H_ + n0 + ncol] = c[r] + bcv;
            }
        }
        __syncthreads();

        // ---- theme MLP: t1 = relu(mh@Ws+bs); theme = sigmoid(t1@Wrs+brs) ----
        for (int e = tid; e < MTILE * HS_; e += THREADS) {
            int m = e / HS_, s = e - m * HS_;
            float acc = bs[s];
            #pragma unroll 4
            for (int hh = 0; hh < H_; ++hh)
                acc += mhS[m * H_ + hh] * Ws[hh * HS_ + s];
            t1S[e] = acc > 0.f ? acc : 0.f;
        }
        __syncthreads();
        for (int e = tid; e < MTILE * H_; e += THREADS) {
            int m = e / H_, o = e - m * H_;
            float acc = brs[o];
            #pragma unroll
            for (int s = 0; s < HS_; ++s)
                acc += t1S[m * HS_ + s] * Wrs[s * H_ + o];
            themeS[e] = 1.f / (1.f + __expf(-acc));
        }
        __syncthreads();

        // ---- output = h + theme*conv; capture last_output at idx ----
        for (int e = tid; e < MTILE * H_; e += THREADS) {
            int m = e / H_, hh = e - m * H_;
            int b = bbase + m;
            float v = hS[e] + themeS[e] * convS[e];
            outSeq[((size_t)b * T_ + t) * H_ + hh] = v;
            if (t == idxS[m]) outLast[(size_t)b * H_ + hh] = v;
        }
        __syncthreads();
    }
}

extern "C" void kernel_launch(void* const* d_in, const int* in_sizes, int n_in,
                              void* d_out, int out_size, void* d_ws, size_t ws_size,
                              hipStream_t stream) {
    (void)in_sizes; (void)n_in; (void)out_size; (void)ws_size;
    const float* x       = (const float*)d_in[0];
    const int*   lengths = (const int*)  d_in[1];
    const float* Wk  = (const float*)d_in[2];
    const float* bk  = (const float*)d_in[3];
    const float* Wr  = (const float*)d_in[4];
    const float* br  = (const float*)d_in[5];
    const float* Ws  = (const float*)d_in[6];
    const float* bs  = (const float*)d_in[7];
    const float* Wrs = (const float*)d_in[8];
    const float* brs = (const float*)d_in[9];
    const float* Wc  = (const float*)d_in[10];
    const float* bc  = (const float*)d_in[11];
    float* out = (float*)d_out;

    h16* WT1 = (h16*)d_ws;                                      // 1542*672  f16
    h16* WT2 = (h16*)((char*)d_ws + (size_t)G_ * KC_ * 2);      // 384*3840  f16

    int n1 = G_ * KC_;
    k_build_wt1<<<(n1 + 255) / 256, 256, 0, stream>>>(Wk, bk, Wr, br, WT1);
    int n2 = H_ * KC2_;
    k_build_wt2<<<(n2 + 255) / 256, 256, 0, stream>>>(Wc, WT2);

    (void)hipFuncSetAttribute((const void*)k_stagenet,
                              hipFuncAttributeMaxDynamicSharedMemorySize, SM_TOTAL);
    k_stagenet<<<B_ / MTILE, THREADS, SM_TOTAL, stream>>>(x, lengths, WT1, WT2,
                                                          Ws, bs, Wrs, brs, bc, out);
}
